// GATEncoder_83803401879708
// MI455X (gfx1250) — compile-verified, matrix operand order checked by reading, not run on hardware
//
#include <hip/hip_runtime.h>
#include <cstddef>

#define NN 100000
#define NE 1600000
#define NEP (NE + NN)

typedef __attribute__((ext_vector_type(16))) _Float16 v16h;
typedef __attribute__((ext_vector_type(8)))  _Float16 v8h;
typedef __attribute__((ext_vector_type(8)))  float    v8f;

// ---- order-preserving float <-> uint key for atomic max on signed floats
__device__ __forceinline__ unsigned fkey(float f) {
  unsigned u = __float_as_uint(f);
  return (u & 0x80000000u) ? ~u : (u | 0x80000000u);
}
__device__ __forceinline__ float fdec(unsigned k) {
  unsigned u = (k & 0x80000000u) ? (k & 0x7fffffffu) : ~k;
  return __uint_as_float(u);
}
__device__ __forceinline__ int esrc(const int* ei, int e) { return e < NE ? ei[e] : e - NE; }
__device__ __forceinline__ int edst(const int* ei, int e) { return e < NE ? ei[NE + e] : e - NE; }

// ---- fold edge-weight matrices with attention vectors:
// M1[d][h] = sum_c W1e[d, h*32+c] * a1e[h, c]   (16x4)
// M2[d]    = sum_c W2e[d, c]      * a2e[c]      (16)
__global__ void fold_kernel(const float* __restrict__ W1e, const float* __restrict__ a1e,
                            const float* __restrict__ W2e, const float* __restrict__ a2e,
                            float* __restrict__ M1, float* __restrict__ M2) {
  int t = threadIdx.x;
  if (t < 64) {
    int d = t >> 2, h = t & 3;
    float s = 0.f;
    for (int c = 0; c < 32; ++c) s += W1e[d * 128 + h * 32 + c] * a1e[h * 32 + c];
    M1[d * 4 + h] = s;
  } else if (t < 80) {
    int d = t - 64;
    float s = 0.f;
    for (int c = 0; c < 32; ++c) s += W2e[d * 32 + c] * a2e[c];
    M2[d] = s;
  }
}

// ---- self-loop mean edge attr: degree + per-dst attr sum
__global__ void deg_sum_kernel(const int* __restrict__ ei, const float* __restrict__ eattr,
                               float* __restrict__ deg, float* __restrict__ sum) {
  int e = blockIdx.x * blockDim.x + threadIdx.x;
  if (e >= NE) return;
  int d = ei[NE + e];
  atomicAdd(&deg[d], 1.0f);
  const float* ea = eattr + (size_t)e * 16;
  float* s = sum + (size_t)d * 16;
  #pragma unroll
  for (int j = 0; j < 16; ++j) atomicAdd(&s[j], ea[j]);
}

__global__ void mean_kernel(const float* __restrict__ deg, float* __restrict__ sum) {
  int i = blockIdx.x * blockDim.x + threadIdx.x;
  if (i >= NN * 16) return;
  sum[i] /= fmaxf(deg[i >> 4], 1.0f);
}

// ---- fp32 -> fp16 conversions (A matrix, and transposed B matrices)
__global__ void cvt_f16_kernel(const float* __restrict__ src, _Float16* __restrict__ dst, int n) {
  int i = blockIdx.x * blockDim.x + threadIdx.x;
  if (i < n) dst[i] = (_Float16)src[i];
}
__global__ void cvt_w_t_kernel(const float* __restrict__ W, _Float16* __restrict__ Wt,
                               int K, int Nc) {
  int i = blockIdx.x * blockDim.x + threadIdx.x;
  if (i >= K * Nc) return;
  int k = i / Nc, n = i - k * Nc;
  Wt[n * K + k] = (_Float16)W[i];  // Wt is [Nc x K] row-major
}

// ---- wave-per-tile WMMA GEMM: C[MxNc] = A[MxK] * Bt[NcxK]^T, f16 in / f32 acc
// Requires M % 16 == 0 (true: 100000 = 6250*16) -> no row guards, EXEC all-ones.
// Each wave: one 16-row A stripe shared across NT column tiles (2 independent
// WMMA accumulation chains interleave to cover the WMMA->WMMA RAW hazard).
// A-fragment (16-bit A 16x32): lanes 0-15 row r, K = {0..7,16..23}; lanes 16-31 K = {8..15,24..31}.
template <int K, int NT>
__global__ void wmma_gemm_kernel(const _Float16* __restrict__ A,
                                 const _Float16* __restrict__ Bt,
                                 float* __restrict__ C, int Nc) {
  const int lane = threadIdx.x;
  const int half = lane >> 4;
  const int l15  = lane & 15;
  const int row  = blockIdx.x * 16 + l15;
  const int klo  = half * 8;
  v8f acc[NT];
  #pragma unroll
  for (int t = 0; t < NT; ++t) acc[t] = (v8f){};
  const _Float16* arow = A + (size_t)row * K + klo;
  #pragma unroll
  for (int k0 = 0; k0 < K; k0 += 32) {
    const v8h* ap = (const v8h*)(arow + k0);
    v8h alo = ap[0];   // K = k0+klo .. +7   (16B contiguous)
    v8h ahi = ap[2];   // K = k0+klo+16 .. +23
    v16h a = __builtin_shufflevector(alo, ahi, 0,1,2,3,4,5,6,7,8,9,10,11,12,13,14,15);
    #pragma unroll
    for (int t = 0; t < NT; ++t) {
      const int col = (blockIdx.y * NT + t) * 16 + l15;
      const v8h* bp = (const v8h*)(Bt + (size_t)col * K + k0 + klo);
      v8h blo = bp[0];
      v8h bhi = bp[2];
      v16h b = __builtin_shufflevector(blo, bhi, 0,1,2,3,4,5,6,7,8,9,10,11,12,13,14,15);
      acc[t] = __builtin_amdgcn_wmma_f32_16x16x32_f16(false, a, false, b, (short)0, acc[t],
                                                      false, false);
    }
  }
  // C/D layout: VGPR v -> M = v + 8*half, N = lane&15
  const int rbase = blockIdx.x * 16 + half * 8;
  #pragma unroll
  for (int t = 0; t < NT; ++t) {
    const int col = (blockIdx.y * NT + t) * 16 + l15;
    #pragma unroll
    for (int v = 0; v < 8; ++v) {
      C[(size_t)(rbase + v) * Nc + col] = acc[t][v];
    }
  }
}

// ---- per-node attention dots: als[n,h] = h[n,h,:]·a_s[h,:], ald likewise
__global__ void attn_node_kernel(const float* __restrict__ hbuf,
                                 const float* __restrict__ a_s, const float* __restrict__ a_d,
                                 float* __restrict__ als, float* __restrict__ ald,
                                 int Hh, int HC) {
  int i = blockIdx.x * blockDim.x + threadIdx.x;
  if (i >= NN * Hh) return;
  int n = i / Hh, hh = i - n * Hh;
  const float* row = hbuf + (size_t)n * HC + hh * 32;
  const float* as = a_s + hh * 32;
  const float* ad = a_d + hh * 32;
  float s = 0.f, d = 0.f;
  #pragma unroll
  for (int c = 0; c < 32; ++c) { float v = row[c]; s += v * as[c]; d += v * ad[c]; }
  als[i] = s;
  ald[i] = d;
}

// ---- per-edge logits + leaky relu + atomic segment max
template <int Hh>
__global__ void edge_logit_kernel(const int* __restrict__ ei,
                                  const float* __restrict__ eattr, const float* __restrict__ mattr,
                                  const float* __restrict__ Me,
                                  const float* __restrict__ als, const float* __restrict__ ald,
                                  float* __restrict__ logits, unsigned* __restrict__ mk) {
  int e = blockIdx.x * blockDim.x + threadIdx.x;
  if (e >= NEP) return;
  int s = esrc(ei, e), d = edst(ei, e);
  const float* ea = (e < NE) ? (eattr + (size_t)e * 16) : (mattr + (size_t)(e - NE) * 16);
  float a[16];
  #pragma unroll
  for (int j = 0; j < 16; ++j) a[j] = ea[j];
  #pragma unroll
  for (int h = 0; h < Hh; ++h) {
    float le = 0.f;
    #pragma unroll
    for (int j = 0; j < 16; ++j) le += a[j] * Me[j * Hh + h];
    float lg = als[s * Hh + h] + ald[d * Hh + h] + le;
    lg = lg > 0.f ? lg : 0.2f * lg;       // leaky relu, slope 0.2
    logits[(size_t)e * Hh + h] = lg;
    atomicMax(&mk[d * Hh + h], fkey(lg));
  }
}

// ---- p = exp(logit - max), atomic segment sum
template <int Hh>
__global__ void edge_exp_kernel(const int* __restrict__ ei, float* __restrict__ logits,
                                const unsigned* __restrict__ mk, float* __restrict__ z) {
  long long i = (long long)blockIdx.x * blockDim.x + threadIdx.x;
  if (i >= (long long)NEP * Hh) return;
  int e = (int)(i / Hh), h = (int)(i - (long long)e * Hh);
  int d = edst(ei, e);
  float p = __expf(logits[i] - fdec(mk[d * Hh + h]));
  logits[i] = p;   // reuse buffer: now holds unnormalized p
  atomicAdd(&z[d * Hh + h], p);
}

// ---- out[dst] += alpha * h[src]; one thread per (edge, 4 channels):
// float4 (b128) gather of h[src], 4x global_atomic_add_f32 scatter.
template <int HC, int Hh>
__global__ void aggregate_kernel(const int* __restrict__ ei, const float* __restrict__ hbuf,
                                 const float* __restrict__ p, const float* __restrict__ z,
                                 float* __restrict__ out, float* __restrict__ alpha_out) {
  constexpr int TPE = HC / 4;  // threads per edge
  long long i = (long long)blockIdx.x * blockDim.x + threadIdx.x;
  if (i >= (long long)NEP * TPE) return;
  int e = (int)(i / TPE), q = (int)(i - (long long)e * TPE);
  int c = q * 4;
  int s = esrc(ei, e), d = edst(ei, e);
  const float* hrow = hbuf + (size_t)s * HC + c;
  __builtin_prefetch(hrow, 0, 0);        // start the gather line early
  int hh = c >> 5;                       // channel -> head (C = 32)
  float alpha = p[(size_t)e * Hh + hh] / z[(size_t)d * Hh + hh];
  float4 hv = *(const float4*)hrow;
  float* orow = out + (size_t)d * HC + c;
  atomicAdd(&orow[0], alpha * hv.x);
  atomicAdd(&orow[1], alpha * hv.y);
  atomicAdd(&orow[2], alpha * hv.z);
  atomicAdd(&orow[3], alpha * hv.w);
  if (alpha_out && q == 0) alpha_out[e] = alpha;
}

// ---- layer-1 epilogue: +bias, ELU, store fp16 (feeds GEMM2 directly)
__global__ void finalize1_kernel(const float* __restrict__ agg, const float* __restrict__ b1,
                                 _Float16* __restrict__ hact) {
  int i = blockIdx.x * blockDim.x + threadIdx.x;
  if (i >= NN * 128) return;
  float v = agg[i] + b1[i & 127];
  v = v > 0.f ? v : __expf(v) - 1.f;     // ELU
  hact[i] = (_Float16)v;
}
// ---- layer-2 epilogue: +bias, ELU, store f32 to output
__global__ void finalize2_kernel(const float* __restrict__ agg, const float* __restrict__ b2,
                                 float* __restrict__ out) {
  int i = blockIdx.x * blockDim.x + threadIdx.x;
  if (i >= NN * 32) return;
  float v = agg[i] + b2[i & 31];
  out[i] = v > 0.f ? v : __expf(v) - 1.f;
}

__global__ void write_edges_kernel(const int* __restrict__ ei, int* __restrict__ out_ei) {
  int e = blockIdx.x * blockDim.x + threadIdx.x;
  if (e >= NEP) return;
  out_ei[e] = esrc(ei, e);
  out_ei[NEP + e] = edst(ei, e);
}

extern "C" void kernel_launch(void* const* d_in, const int* in_sizes, int n_in,
                              void* d_out, int out_size, void* d_ws, size_t ws_size,
                              hipStream_t stream) {
  const float* x    = (const float*)d_in[0];
  const int*   ei   = (const int*)d_in[1];
  const float* eatt = (const float*)d_in[2];
  const float* W1   = (const float*)d_in[3];
  const float* W1e  = (const float*)d_in[4];
  const float* a1s  = (const float*)d_in[5];
  const float* a1d  = (const float*)d_in[6];
  const float* a1e  = (const float*)d_in[7];
  const float* b1   = (const float*)d_in[8];
  const float* W2   = (const float*)d_in[9];
  const float* W2e  = (const float*)d_in[10];
  const float* a2s  = (const float*)d_in[11];
  const float* a2d  = (const float*)d_in[12];
  const float* a2e  = (const float*)d_in[13];
  const float* b2   = (const float*)d_in[14];
  (void)in_sizes; (void)n_in; (void)out_size; (void)ws_size;

  char* w = (char*)d_ws;
  auto take = [&](size_t bytes) { char* p = w; w += (bytes + 255) & ~(size_t)255; return p; };
  float*    M1    = (float*)take(64 * 4);
  float*    M2    = (float*)take(16 * 4);
  float*    deg   = (float*)take((size_t)NN * 4);
  float*    mattr = (float*)take((size_t)NN * 16 * 4);
  _Float16* x16   = (_Float16*)take((size_t)NN * 128 * 2);
  _Float16* W1t   = (_Float16*)take(128 * 128 * 2);
  _Float16* W2t   = (_Float16*)take(32 * 128 * 2);
  float*    h1    = (float*)take((size_t)NN * 128 * 4);
  _Float16* h1a   = (_Float16*)take((size_t)NN * 128 * 2);
  float*    h2    = (float*)take((size_t)NN * 32 * 4);
  float*    als1  = (float*)take((size_t)NN * 4 * 4);
  float*    ald1  = (float*)take((size_t)NN * 4 * 4);
  float*    als2  = (float*)take((size_t)NN * 4);
  float*    ald2  = (float*)take((size_t)NN * 4);
  unsigned* m1    = (unsigned*)take((size_t)NN * 4 * 4);
  float*    z1    = (float*)take((size_t)NN * 4 * 4);
  unsigned* m2    = (unsigned*)take((size_t)NN * 4);
  float*    z2    = (float*)take((size_t)NN * 4);
  float*    lg1   = (float*)take((size_t)NEP * 4 * 4);
  float*    lg2   = (float*)take((size_t)NEP * 4);
  float*    agg1  = (float*)take((size_t)NN * 128 * 4);
  float*    agg2  = (float*)take((size_t)NN * 32 * 4);

  // output tuple layout: h2 [NN*32] | edge_index_sl [2*NEP] (int bits) | alpha2 [NEP]
  float* out_h2    = (float*)d_out;
  int*   out_ei    = (int*)((float*)d_out + (size_t)NN * 32);
  float* out_alpha = (float*)d_out + (size_t)NN * 32 + (size_t)2 * NEP;

  hipMemsetAsync(deg,   0, (size_t)NN * 4, stream);
  hipMemsetAsync(mattr, 0, (size_t)NN * 16 * 4, stream);
  hipMemsetAsync(m1,    0, (size_t)NN * 4 * 4, stream);
  hipMemsetAsync(z1,    0, (size_t)NN * 4 * 4, stream);
  hipMemsetAsync(m2,    0, (size_t)NN * 4, stream);
  hipMemsetAsync(z2,    0, (size_t)NN * 4, stream);
  hipMemsetAsync(agg1,  0, (size_t)NN * 128 * 4, stream);
  hipMemsetAsync(agg2,  0, (size_t)NN * 32 * 4, stream);

  auto g = [](long long n, int b) { return dim3((unsigned)((n + b - 1) / b)); };

  fold_kernel<<<1, 128, 0, stream>>>(W1e, a1e, W2e, a2e, M1, M2);
  deg_sum_kernel<<<g(NE, 256), 256, 0, stream>>>(ei, eatt, deg, mattr);
  mean_kernel<<<g((long long)NN * 16, 256), 256, 0, stream>>>(deg, mattr);
  cvt_f16_kernel<<<g((long long)NN * 128, 256), 256, 0, stream>>>(x, x16, NN * 128);
  cvt_w_t_kernel<<<g(128 * 128, 256), 256, 0, stream>>>(W1, W1t, 128, 128);
  cvt_w_t_kernel<<<g(128 * 32, 256), 256, 0, stream>>>(W2, W2t, 128, 32);

  // ---- layer 1  (M=100000 -> 6250 row tiles; Nc=128 -> 4 blocks of NT=2 col tiles)
  wmma_gemm_kernel<128, 2><<<dim3(NN / 16, 128 / 32), 32, 0, stream>>>(x16, W1t, h1, 128);
  attn_node_kernel<<<g((long long)NN * 4, 256), 256, 0, stream>>>(h1, a1s, a1d, als1, ald1, 4, 128);
  edge_logit_kernel<4><<<g(NEP, 256), 256, 0, stream>>>(ei, eatt, mattr, M1, als1, ald1, lg1, m1);
  edge_exp_kernel<4><<<g((long long)NEP * 4, 256), 256, 0, stream>>>(ei, lg1, m1, z1);
  aggregate_kernel<128, 4><<<g((long long)NEP * 32, 256), 256, 0, stream>>>(ei, h1, lg1, z1, agg1, nullptr);
  finalize1_kernel<<<g((long long)NN * 128, 256), 256, 0, stream>>>(agg1, b1, h1a);

  // ---- layer 2  (Nc=32 -> 1 block of NT=2 col tiles)
  wmma_gemm_kernel<128, 2><<<dim3(NN / 16, 1), 32, 0, stream>>>(h1a, W2t, h2, 32);
  attn_node_kernel<<<g(NN, 256), 256, 0, stream>>>(h2, a2s, a2d, als2, ald2, 1, 32);
  edge_logit_kernel<1><<<g(NEP, 256), 256, 0, stream>>>(ei, eatt, mattr, M2, als2, ald2, lg2, m2);
  edge_exp_kernel<1><<<g(NEP, 256), 256, 0, stream>>>(ei, lg2, m2, z2);
  aggregate_kernel<32, 1><<<g((long long)NEP * 8, 256), 256, 0, stream>>>(ei, h2, lg2, z2, agg2, out_alpha);
  finalize2_kernel<<<g((long long)NN * 32, 256), 256, 0, stream>>>(agg2, b2, out_h2);
  write_edges_kernel<<<g(NEP, 256), 256, 0, stream>>>(ei, out_ei);
}
